// HRGRAgent_1992864825929
// MI455X (gfx1250) — matrix operand order repaired and checked by reading, not compiled, not run on hardware
//
#include <hip/hip_runtime.h>
#include <hip/hip_bf16.h>
#include <hip/hip_fp16.h>

// ---------------- dims ----------------
constexpr int nB = 16;       // batch
constexpr int nC = 768;      // encoder dim
constexpr int nP = 288;      // 2*144 spatial positions
constexpr int nV = 20000;    // vocab
constexpr int nE = 512;      // embed
constexpr int nD = 512;      // decoder dim
constexpr int nA = 512;      // attention dim
constexpr int nS = 8;        // sentences
constexpr int nW = 20;       // words
constexpr int nR = nB * nS * nW;   // 2560 word-step rows
constexpr int nM_ATT = nB * nP;    // 4608 rows for att1 GEMM

typedef __attribute__((ext_vector_type(16))) _Float16 v16h;
typedef __attribute__((ext_vector_type(8)))  _Float16 v8h;
typedef __attribute__((ext_vector_type(8)))  float    v8f;

__device__ __forceinline__ void sched_fence() {
#if __has_builtin(__builtin_amdgcn_sched_barrier)
  __builtin_amdgcn_sched_barrier(0);   // keep loads grouped ahead of WMMAs
#endif
}

// Fragment packing without element shuffles: loads land directly in halves.
union AFrag { v16h v; struct { v8h lo; v8h hi; } s; };

__device__ __forceinline__ v16h load_afrag(const _Float16* p) {
  // A lane layout: 16 contiguous halves at k + 16*(lane>>4)
  AFrag u;
  u.s.lo = *(const v8h*)(p);
  u.s.hi = *(const v8h*)(p + 8);
  return u.v;
}
__device__ __forceinline__ v16h load_bfrag(const _Float16* p) {
  // B lane layout: 16 contiguous halves at k + 16*(lane>>4)  -> one 32B load
  return *(const v16h*)(p);
}

// =====================================================================
// WMMA GEMM:  out[M x N] = A[M x K](f16) * W[N x K](f16)^T + bias
// One wave computes a 16(M) x (16*NTILES)(N) strip, 4 waves/block over M.
// NTILES is a template param so the K-loop is branch-free (EXEC all-ones,
// accumulators pinned). K must be a multiple of 64. N tiles are 16-aligned.
// outTrans=1 stores out[n*ldo + m] (used for att1 transpose).
// =====================================================================
template <int NTILES>
__global__ __launch_bounds__(128) void gemm_wmma_kernel(
    const _Float16* __restrict__ Am, int lda,
    const _Float16* __restrict__ Wm, int ldw,
    const float* __restrict__ bias,
    float* __restrict__ out, int ldo, int outTrans,
    int M, int K, int nbase0)
{
  const int lane = threadIdx.x & 31;
  const int wave = threadIdx.x >> 5;
  const int mtile = blockIdx.x * 4 + wave;
  if (mtile * 16 >= M) return;
  const int nbase = nbase0 + blockIdx.y * (16 * NTILES);
  const int mrow  = mtile * 16 + (lane & 15);
  const int ncol  = (lane & 15);
  const int ah    = (lane >> 4) * 16;   // A k sub-offset
  const int bh    = (lane >> 4) * 16;   // B k sub-offset

  const _Float16* aptr = Am + (size_t)mrow * lda + ah;
  const _Float16* bptr[NTILES];
#pragma unroll
  for (int t = 0; t < NTILES; ++t)
    bptr[t] = Wm + (size_t)(nbase + t * 16 + ncol) * ldw + bh;

  const v8f zero8 = {0.f,0.f,0.f,0.f,0.f,0.f,0.f,0.f};
  v8f acc[NTILES];
#pragma unroll
  for (int t = 0; t < NTILES; ++t) acc[t] = zero8;

  // K step 64: issue ALL fragment loads (clustered clauses, all in flight),
  // then fence the scheduler, then run 2*NTILES WMMAs back-to-back. One
  // s_wait_loadcnt exposure per iteration instead of one per WMMA.
  for (int k0 = 0; k0 < K; k0 += 64) {
    v16h a0 = load_afrag(aptr + k0);
    v16h a1 = load_afrag(aptr + k0 + 32);
    v16h b0[NTILES], b1[NTILES];
#pragma unroll
    for (int t = 0; t < NTILES; ++t) b0[t] = load_bfrag(bptr[t] + k0);
#pragma unroll
    for (int t = 0; t < NTILES; ++t) b1[t] = load_bfrag(bptr[t] + k0 + 32);
    sched_fence();
#pragma unroll
    for (int t = 0; t < NTILES; ++t)
      acc[t] = __builtin_amdgcn_wmma_f32_16x16x32_f16(
          false, a0, false, b0[t], (short)0, acc[t], false, false);
#pragma unroll
    for (int t = 0; t < NTILES; ++t)
      acc[t] = __builtin_amdgcn_wmma_f32_16x16x32_f16(
          false, a1, false, b1[t], (short)0, acc[t], false, false);
  }

#pragma unroll
  for (int t = 0; t < NTILES; ++t) {
    const int n = nbase + t * 16 + ncol;
    const float bv = bias ? bias[n] : 0.f;
#pragma unroll
    for (int j = 0; j < 8; ++j) {
      const int m = mtile * 16 + j + ((lane >> 4) * 8);
      const float v = acc[t][j] + bv;
      if (outTrans) out[(size_t)n * ldo + m] = v;
      else          out[(size_t)m * ldo + n] = v;
    }
  }
}

static void launch_gemm(const _Float16* A, int lda, const _Float16* W, int ldw,
                        const float* bias, float* out, int ldo, int outTrans,
                        int M, int N, int K, hipStream_t stream)
{
  const int mblocks = (M / 16 + 3) / 4;
  const int full = N / 64;          // 64-wide strips
  const int rem  = (N % 64) / 16;   // remaining 16-wide tiles (N % 16 == 0)
  if (full > 0)
    gemm_wmma_kernel<4><<<dim3(mblocks, full), 128, 0, stream>>>(
        A, lda, W, ldw, bias, out, ldo, outTrans, M, K, 0);
  if (rem == 1)
    gemm_wmma_kernel<1><<<dim3(mblocks, 1), 128, 0, stream>>>(
        A, lda, W, ldw, bias, out, ldo, outTrans, M, K, full * 64);
  else if (rem == 2)
    gemm_wmma_kernel<2><<<dim3(mblocks, 1), 128, 0, stream>>>(
        A, lda, W, ldw, bias, out, ldo, outTrans, M, K, full * 64);
  else if (rem == 3)
    gemm_wmma_kernel<3><<<dim3(mblocks, 1), 128, 0, stream>>>(
        A, lda, W, ldw, bias, out, ldo, outTrans, M, K, full * 64);
}

// ---------------- small helper kernels ----------------
__global__ void f32_to_f16_kernel(const float* __restrict__ in,
                                  _Float16* __restrict__ out, int n) {
  int i = blockIdx.x * blockDim.x + threadIdx.x;
  if (i < n) out[i] = (_Float16)in[i];
}

// out[c*J + j] = in[j*ld + off + c]   (transposed f16 weight copy, coalesced writes)
__global__ void transpose_f16_kernel(const float* __restrict__ in,
                                     _Float16* __restrict__ out,
                                     int J, int Kc, int ld, int off) {
  int i = blockIdx.x * blockDim.x + threadIdx.x;
  if (i >= J * Kc) return;
  int j = i % J, c = i / J;
  out[i] = (_Float16)in[(size_t)j * ld + off + c];
}

// spat_h[(b*288+p)*768 + c] = concat(spatial_new, spatial_old) in f16
__global__ void spat_kernel(const float* __restrict__ sn, const float* __restrict__ so,
                            _Float16* __restrict__ spat_h) {
  int i = blockIdx.x * blockDim.x + threadIdx.x;
  if (i >= nM_ATT * nC) return;
  int row = i / nC, c = i % nC;
  int b = row / nP, p = row % nP;
  float v = (p < 144) ? sn[((size_t)b * 144 + p) * nC + c]
                      : so[((size_t)b * 144 + (p - 144)) * nC + c];
  spat_h[i] = (_Float16)v;
}

// gcomb[b][k] : k<768 -> mean(spatial_new), else mean(spatial_old)
__global__ void gmean_kernel(const float* __restrict__ sn, const float* __restrict__ so,
                             float* __restrict__ gcomb) {
  int i = blockIdx.x * blockDim.x + threadIdx.x;
  if (i >= nB * 2 * nC) return;
  int b = i / (2 * nC), k = i % (2 * nC);
  const float* src = (k < nC) ? sn : so;
  int c = (k < nC) ? k : k - nC;
  float s = 0.f;
  for (int p = 0; p < 144; ++p) s += src[((size_t)b * 144 + p) * nC + c];
  gcomb[i] = s * (1.0f / 144.0f);
}

// g_fused = gcomb @ W_gfuse^T + b ; h_s0 = tanh(gcomb @ W_init^T + b)
__global__ void fuse_kernel(const float* __restrict__ gcomb,
                            const float* __restrict__ W_gfuse, const float* __restrict__ b_gfuse,
                            const float* __restrict__ W_init,  const float* __restrict__ b_init,
                            float* __restrict__ g_fused, float* __restrict__ h_s0) {
  int i = blockIdx.x * blockDim.x + threadIdx.x;
  if (i >= nB * (nC + nD)) return;
  int b = i / (nC + nD), j = i % (nC + nD);
  const float* g = gcomb + (size_t)b * (2 * nC);
  if (j < nC) {
    float acc = b_gfuse[j];
    const float* wr = W_gfuse + (size_t)j * (2 * nC);
    for (int k = 0; k < 2 * nC; ++k) acc += g[k] * wr[k];
    g_fused[b * nC + j] = acc;
  } else {
    int jj = j - nC;
    float acc = b_init[jj];
    const float* wr = W_init + (size_t)jj * (2 * nC);
    for (int k = 0; k < 2 * nC; ++k) acc += g[k] * wr[k];
    h_s0[b * nD + jj] = tanhf(acc);
  }
}

// teacher-forced prev-word embedding gather -> f16 rows ordered r=((b*S+s)*W+w)
__global__ void emb_gather_kernel(const float* __restrict__ W_emb,
                                  const int* __restrict__ tw,
                                  _Float16* __restrict__ emb_h) {
  int i = blockIdx.x * blockDim.x + threadIdx.x;
  if (i >= nR * nE) return;
  int r = i >> 9, e = i & 511;
  int b = r / (nS * nW);
  int rem = r % (nS * nW);
  int s = rem / nW, w = rem % nW;
  int wid = (w == 0) ? 1 : tw[(b * nS + s) * nW + (w - 1)];
  emb_h[i] = (_Float16)W_emb[(size_t)wid * nE + e];
}

__device__ __forceinline__ float sigmf(float x) { return 1.f / (1.f + __expf(-x)); }

// ---------------- sentence GRU chain: 1 block per batch element ----------------
__global__ __launch_bounds__(256) void sent_kernel(
    const float* __restrict__ g_fused, const float* __restrict__ h_s0,
    const _Float16* __restrict__ WsihT, const _Float16* __restrict__ WshhT,
    const float* __restrict__ bs_ih, const float* __restrict__ bs_hh,
    const float* __restrict__ W_pol, const float* __restrict__ b_pol,
    const float* __restrict__ W_stop, const float* __restrict__ b_stop,
    float* __restrict__ h_s_all, float* __restrict__ out_pol, float* __restrict__ out_stp)
{
  __shared__ float h[nD], gf[nC], gi[3 * nD], gh[3 * nD];
  const int b = blockIdx.x, t = threadIdx.x;
  for (int j = t; j < nD; j += 256) h[j] = h_s0[b * nD + j];
  for (int j = t; j < nC; j += 256) gf[j] = g_fused[b * nC + j];
  __syncthreads();
  for (int s = 0; s < nS; ++s) {
    float ai[6], ah[6];
#pragma unroll
    for (int q = 0; q < 6; ++q) { ai[q] = bs_ih[t + 256 * q]; ah[q] = bs_hh[t + 256 * q]; }
    for (int k = 0; k < nC; ++k) {
      float gv = gf[k];
      const _Float16* wr = WsihT + (size_t)k * (3 * nD) + t;
#pragma unroll
      for (int q = 0; q < 6; ++q) ai[q] += gv * (float)wr[256 * q];
    }
    for (int d = 0; d < nD; ++d) {
      float hv = h[d];
      const _Float16* wr = WshhT + (size_t)d * (3 * nD) + t;
#pragma unroll
      for (int q = 0; q < 6; ++q) ah[q] += hv * (float)wr[256 * q];
    }
#pragma unroll
    for (int q = 0; q < 6; ++q) { gi[t + 256 * q] = ai[q]; gh[t + 256 * q] = ah[q]; }
    __syncthreads();
    for (int j = t; j < nD; j += 256) {
      float rr = sigmf(gi[j] + gh[j]);
      float zz = sigmf(gi[nD + j] + gh[nD + j]);
      float nn = tanhf(gi[2 * nD + j] + rr * gh[2 * nD + j]);
      float hn = (1.f - zz) * nn + zz * h[j];
      h[j] = hn;
      h_s_all[((size_t)s * nB + b) * nD + j] = hn;
    }
    __syncthreads();
    if (t < 11) {
      float acc = b_pol[t];
      for (int d = 0; d < nD; ++d) acc += h[d] * W_pol[t * nD + d];
      out_pol[(b * nS + s) * 11 + t] = acc;
    } else if (t == 11) {
      float acc = b_stop[0];
      for (int d = 0; d < nD; ++d) acc += h[d] * W_stop[d];
      out_stp[b * nS + s] = acc;
    }
    __syncthreads();
  }
}

// ---------------- word chains: grid (B, S), 20 sequential steps each ----------------
__global__ __launch_bounds__(256) void word_kernel(
    const float* __restrict__ h_s_all, const float* __restrict__ att1T,
    const _Float16* __restrict__ spat_h, const float* __restrict__ gi_emb,
    const _Float16* __restrict__ W_decT, const float* __restrict__ b_dec,
    const float* __restrict__ W_full, const float* __restrict__ b_full,
    const _Float16* __restrict__ WwihcT, const _Float16* __restrict__ WwhhT,
    const float* __restrict__ bw_hh, _Float16* __restrict__ h_all_h)
{
  __shared__ float h[nD], att2[nA], ebuf[nP], ctx[nC], gi[3 * nD], gh[3 * nD], red[256];
  const int b = blockIdx.x, s = blockIdx.y, t = threadIdx.x;
  for (int j = t; j < nD; j += 256) h[j] = h_s_all[((size_t)s * nB + b) * nD + j];
  __syncthreads();

  for (int w = 0; w < nW; ++w) {
    const int r = (b * nS + s) * nW + w;
    // att2 = h @ W_dec^T + b_dec   (2 outputs/thread, coalesced f16 reads)
    {
      float a0 = b_dec[t], a1 = b_dec[t + 256];
      for (int d = 0; d < nD; ++d) {
        float hv = h[d];
        a0 += hv * (float)W_decT[d * nA + t];
        a1 += hv * (float)W_decT[d * nA + t + 256];
      }
      att2[t] = a0; att2[t + 256] = a1;
    }
    __syncthreads();
    // e[p] = relu(att1 + att2) . W_full + b_full   (att1T is [a][b*288+p])
    for (int p = t; p < nP; p += 256) {
      float acc = b_full[0];
      for (int a = 0; a < nA; ++a) {
        float v = att1T[(size_t)a * nM_ATT + b * nP + p] + att2[a];
        acc += fmaxf(v, 0.f) * W_full[a];
      }
      ebuf[p] = acc;
    }
    __syncthreads();
    // softmax over 288
    float pm = -1e30f;
    for (int p = t; p < nP; p += 256) pm = fmaxf(pm, ebuf[p]);
    red[t] = pm; __syncthreads();
    for (int o = 128; o > 0; o >>= 1) { if (t < o) red[t] = fmaxf(red[t], red[t + o]); __syncthreads(); }
    float mx = red[0]; __syncthreads();
    float ps = 0.f;
    for (int p = t; p < nP; p += 256) { float v = __expf(ebuf[p] - mx); ebuf[p] = v; ps += v; }
    red[t] = ps; __syncthreads();
    for (int o = 128; o > 0; o >>= 1) { if (t < o) red[t] += red[t + o]; __syncthreads(); }
    float sinv = 1.f / red[0]; __syncthreads();
    for (int p = t; p < nP; p += 256) ebuf[p] *= sinv;
    __syncthreads();
    // ctx[c] = sum_p alpha[p]*spatial[b][p][c]  (3 outputs/thread)
    {
      float c0 = 0.f, c1 = 0.f, c2 = 0.f;
      for (int p = 0; p < nP; ++p) {
        float al = ebuf[p];
        const _Float16* sp = spat_h + ((size_t)b * nP + p) * nC;
        c0 += al * (float)sp[t];
        c1 += al * (float)sp[t + 256];
        c2 += al * (float)sp[t + 512];
      }
      ctx[t] = c0; ctx[t + 256] = c1; ctx[t + 512] = c2;
    }
    __syncthreads();
    // gi = gi_emb (precomputed emb part + bias) + ctx @ Wwih_ctx^T ; gh = h @ Ww_hh^T + b
    {
      float ai[6], ag[6];
#pragma unroll
      for (int q = 0; q < 6; ++q) {
        ai[q] = gi_emb[(size_t)r * (3 * nD) + t + 256 * q];
        ag[q] = bw_hh[t + 256 * q];
      }
      for (int c = 0; c < nC; ++c) {
        float cv = ctx[c];
        const _Float16* wr = WwihcT + (size_t)c * (3 * nD) + t;
#pragma unroll
        for (int q = 0; q < 6; ++q) ai[q] += cv * (float)wr[256 * q];
      }
      for (int d = 0; d < nD; ++d) {
        float hv = h[d];
        const _Float16* wr = WwhhT + (size_t)d * (3 * nD) + t;
#pragma unroll
        for (int q = 0; q < 6; ++q) ag[q] += hv * (float)wr[256 * q];
      }
#pragma unroll
      for (int q = 0; q < 6; ++q) { gi[t + 256 * q] = ai[q]; gh[t + 256 * q] = ag[q]; }
    }
    __syncthreads();
    // GRU update + emit f16 state row for the logits GEMM
    for (int j = t; j < nD; j += 256) {
      float rr = sigmf(gi[j] + gh[j]);
      float zz = sigmf(gi[nD + j] + gh[nD + j]);
      float nn = tanhf(gi[2 * nD + j] + rr * gh[2 * nD + j]);
      float hn = (1.f - zz) * nn + zz * h[j];
      h[j] = hn;
      h_all_h[(size_t)r * nD + j] = (_Float16)hn;
    }
    __syncthreads();
  }
}

// =====================================================================
extern "C" void kernel_launch(void* const* d_in, const int* in_sizes, int n_in,
                              void* d_out, int out_size, void* d_ws, size_t ws_size,
                              hipStream_t stream) {
  const float* spatial_new = (const float*)d_in[0];
  const float* spatial_old = (const float*)d_in[1];
  const float* W_emb   = (const float*)d_in[2];
  const float* W_gfuse = (const float*)d_in[3];
  const float* b_gfuse = (const float*)d_in[4];
  const float* W_init  = (const float*)d_in[5];
  const float* b_init  = (const float*)d_in[6];
  const float* Ws_ih   = (const float*)d_in[7];
  const float* Ws_hh   = (const float*)d_in[8];
  const float* bs_ih   = (const float*)d_in[9];
  const float* bs_hh   = (const float*)d_in[10];
  const float* W_pol   = (const float*)d_in[11];
  const float* b_pol   = (const float*)d_in[12];
  const float* W_stop  = (const float*)d_in[13];
  const float* b_stop  = (const float*)d_in[14];
  const float* W_enc   = (const float*)d_in[15];
  const float* b_enc   = (const float*)d_in[16];
  const float* W_dec   = (const float*)d_in[17];
  const float* b_dec   = (const float*)d_in[18];
  const float* W_full  = (const float*)d_in[19];
  const float* b_full  = (const float*)d_in[20];
  const float* Ww_ih   = (const float*)d_in[21];
  const float* Ww_hh   = (const float*)d_in[22];
  const float* bw_ih   = (const float*)d_in[23];
  const float* bw_hh   = (const float*)d_in[24];
  const float* W_fc    = (const float*)d_in[25];
  const float* b_fc    = (const float*)d_in[26];
  const int*   tw      = (const int*)d_in[27];

  float* out = (float*)d_out;
  float* out_pol = out;                          // (B,S,11)
  float* out_stp = out + nB * nS * 11;           // (B,S,1)
  float* out_wl  = out + nB * nS * 11 + nB * nS; // (B,S,W,V)

  // ---- workspace carve ----
  char* ws = (char*)d_ws;
  size_t off = 0;
  auto carve = [&](size_t bytes) -> void* {
    off = (off + 255) & ~(size_t)255;
    void* p = ws + off; off += bytes; return p;
  };
  float* gcomb   = (float*)carve((size_t)nB * 2 * nC * 4);
  float* g_fused = (float*)carve((size_t)nB * nC * 4);
  float* h_s0    = (float*)carve((size_t)nB * nD * 4);
  float* h_s_all = (float*)carve((size_t)nS * nB * nD * 4);
  float* att1T   = (float*)carve((size_t)nA * nM_ATT * 4);
  float* gi_emb  = (float*)carve((size_t)nR * 3 * nD * 4);
  _Float16* W_fc_h   = (_Float16*)carve((size_t)nV * nD * 2);
  _Float16* W_enc_h  = (_Float16*)carve((size_t)nA * nC * 2);
  _Float16* Ww_ih_h  = (_Float16*)carve((size_t)(3 * nD) * (nE + nC) * 2);
  _Float16* spat_h   = (_Float16*)carve((size_t)nM_ATT * nC * 2);
  _Float16* emb_h    = (_Float16*)carve((size_t)nR * nE * 2);
  _Float16* h_all_h  = (_Float16*)carve((size_t)nR * nD * 2);
  _Float16* W_decT   = (_Float16*)carve((size_t)nD * nA * 2);
  _Float16* WwhhT    = (_Float16*)carve((size_t)nD * 3 * nD * 2);
  _Float16* WwihcT   = (_Float16*)carve((size_t)nC * 3 * nD * 2);
  _Float16* WsihT    = (_Float16*)carve((size_t)nC * 3 * nD * 2);
  _Float16* WshhT    = (_Float16*)carve((size_t)nD * 3 * nD * 2);

  const int T = 256;
  auto nb = [&](long n) { return dim3((unsigned)((n + T - 1) / T)); };

  // ---- precision / layout prep ----
  f32_to_f16_kernel<<<nb((long)nV * nD), T, 0, stream>>>(W_fc, W_fc_h, nV * nD);
  f32_to_f16_kernel<<<nb((long)nA * nC), T, 0, stream>>>(W_enc, W_enc_h, nA * nC);
  f32_to_f16_kernel<<<nb((long)3 * nD * (nE + nC)), T, 0, stream>>>(Ww_ih, Ww_ih_h, 3 * nD * (nE + nC));
  transpose_f16_kernel<<<nb((long)nA * nD), T, 0, stream>>>(W_dec, W_decT, nA, nD, nD, 0);
  transpose_f16_kernel<<<nb((long)3 * nD * nD), T, 0, stream>>>(Ww_hh, WwhhT, 3 * nD, nD, nD, 0);
  transpose_f16_kernel<<<nb((long)3 * nD * nC), T, 0, stream>>>(Ww_ih, WwihcT, 3 * nD, nC, nE + nC, nE);
  transpose_f16_kernel<<<nb((long)3 * nD * nC), T, 0, stream>>>(Ws_ih, WsihT, 3 * nD, nC, nC, 0);
  transpose_f16_kernel<<<nb((long)3 * nD * nD), T, 0, stream>>>(Ws_hh, WshhT, 3 * nD, nD, nD, 0);
  spat_kernel<<<nb((long)nM_ATT * nC), T, 0, stream>>>(spatial_new, spatial_old, spat_h);

  // ---- globals, fuse, init state, embeddings ----
  gmean_kernel<<<nb((long)nB * 2 * nC), T, 0, stream>>>(spatial_new, spatial_old, gcomb);
  fuse_kernel<<<nb((long)nB * (nC + nD)), T, 0, stream>>>(gcomb, W_gfuse, b_gfuse, W_init, b_init,
                                                          g_fused, h_s0);
  emb_gather_kernel<<<nb((long)nR * nE), T, 0, stream>>>(W_emb, tw, emb_h);

  // ---- hoisted WMMA GEMMs ----
  // att1T[a][b*288+p] = (spatial @ W_enc^T + b_enc)^T : M=4608,N=512,K=768
  launch_gemm(spat_h, nC, W_enc_h, nC, b_enc, att1T, nM_ATT, 1, nM_ATT, nA, nC, stream);
  // gi_emb = emb @ Ww_ih[:, :512]^T + bw_ih : M=2560,N=1536,K=512 (ldw=1280)
  launch_gemm(emb_h, nE, Ww_ih_h, nE + nC, bw_ih, gi_emb, 3 * nD, 0, nR, 3 * nD, nE, stream);

  // ---- recurrent chains ----
  sent_kernel<<<dim3(nB), 256, 0, stream>>>(g_fused, h_s0, WsihT, WshhT, bs_ih, bs_hh,
                                            W_pol, b_pol, W_stop, b_stop,
                                            h_s_all, out_pol, out_stp);
  word_kernel<<<dim3(nB, nS), 256, 0, stream>>>(h_s_all, att1T, spat_h, gi_emb,
                                                W_decT, b_dec, W_full, b_full,
                                                WwihcT, WwhhT, bw_hh, h_all_h);

  // ---- logits: wl = H @ W_fc^T + b_fc : M=2560, N=20000, K=512 ----
  launch_gemm(h_all_h, nD, W_fc_h, nD, b_fc, out_wl, nV, 0, nR, nV, nD, stream);

  (void)in_sizes; (void)n_in; (void)out_size; (void)ws_size;
}
